// ConformerAttentionMetal_50723563766132
// MI455X (gfx1250) — matrix operand order repaired and checked by reading, not compiled, not run on hardware
//
#include <hip/hip_runtime.h>
#include <stdint.h>

// ---------------------------------------------------------------------------
// Problem constants
// ---------------------------------------------------------------------------
#define BATCH   2
#define SEQ     2048
#define HID     1024
#define NHEADS  16
#define HDIM    64
#define MTOT    (BATCH * SEQ)          // 4096 rows for all dense GEMMs
#define QK_SCALE 0.125f                // 1/sqrt(64)
#define LOG2E   1.44269504088896340736f
#define NEG_BIG (-3.0e38f)
#define NEG_THR (-1.0e37f)

// ---------------------------------------------------------------------------
// WMMA types / helpers (CDNA5 gfx1250, wave32)
// ---------------------------------------------------------------------------
typedef __attribute__((ext_vector_type(16))) __bf16 v16bf;
typedef __attribute__((ext_vector_type(8)))  float  v8f;

union Frag16 { v16bf v; uint4 u[2]; };

__device__ __forceinline__ v8f v8f_zero() {
    v8f z;
#pragma unroll
    for (int i = 0; i < 8; ++i) z[i] = 0.0f;
    return z;
}

__device__ __forceinline__ v8f wmma_bf16(v16bf a, v16bf b, v8f c) {
    // D = A(16x32 bf16) * B(32x16 bf16) + C(16x16 f32)
    return __builtin_amdgcn_wmma_f32_16x16x32_bf16(
        /*neg_a=*/false, a, /*neg_b=*/false, b,
        /*c_mod=*/(short)0, c, /*reuse_a=*/false, /*reuse_b=*/false);
}

// float -> bf16 raw bits via native convert (v_cvt bf16, RNE)
__device__ __forceinline__ unsigned short f2bf(float f) {
    __bf16 b = (__bf16)f;
    union { __bf16 b; unsigned short s; } u;
    u.b = b;
    return u.s;
}

// ---------------------------------------------------------------------------
// DPP16 butterfly reductions across the 16 lanes of a half-wave (a DPP "row").
// No LDS, no dscnt waits — pure VALU, unlike ds_bpermute-based __shfl_xor.
// ---------------------------------------------------------------------------
template <int CTRL>
__device__ __forceinline__ float dpp_mov_f32(float x) {
    return __int_as_float(
        __builtin_amdgcn_mov_dpp(__float_as_int(x), CTRL, 0xF, 0xF, true));
}

__device__ __forceinline__ float row_max16(float v) {
    v = fmaxf(v, dpp_mov_f32<0xB1>(v));    // quad_perm [1,0,3,2]  : xor 1
    v = fmaxf(v, dpp_mov_f32<0x4E>(v));    // quad_perm [2,3,0,1]  : xor 2
    v = fmaxf(v, dpp_mov_f32<0x141>(v));   // row_half_mirror      : cross-quad
    v = fmaxf(v, dpp_mov_f32<0x140>(v));   // row_mirror           : cross-half
    return v;
}

__device__ __forceinline__ float row_sum16(float v) {
    v += dpp_mov_f32<0xB1>(v);
    v += dpp_mov_f32<0x4E>(v);
    v += dpp_mov_f32<0x141>(v);
    v += dpp_mov_f32<0x140>(v);
    return v;
}

// ---------------------------------------------------------------------------
// Elementwise f32 -> bf16 convert (4 elements / thread)
// ---------------------------------------------------------------------------
__global__ __launch_bounds__(256)
void cvt_bf16_kernel(const float* __restrict__ src, unsigned short* __restrict__ dst, int n4) {
    int i = blockIdx.x * blockDim.x + threadIdx.x;
    if (i >= n4) return;
    float4 f = ((const float4*)src)[i];
    ushort4 o;
    o.x = f2bf(f.x); o.y = f2bf(f.y); o.z = f2bf(f.z); o.w = f2bf(f.w);
    ((ushort4*)dst)[i] = o;
}

// ---------------------------------------------------------------------------
// pos_bias[b][h][s] = dot(pos_emb[b][s][:], w_pos[h][:])   (tiny: 0.13 GF)
// ---------------------------------------------------------------------------
__global__ __launch_bounds__(256)
void pos_bias_kernel(const float* __restrict__ pos_emb, const float* __restrict__ w_pos,
                     float* __restrict__ pb) {
    int idx = blockIdx.x * blockDim.x + threadIdx.x;     // B*S*NHEADS threads
    int h   = idx & (NHEADS - 1);
    int row = idx >> 4;                                   // 0 .. B*S-1
    const float4* pe = (const float4*)(pos_emb + (size_t)row * HID);
    const float4* wp = (const float4*)(w_pos   + (size_t)h   * HID);
    float s = 0.0f;
#pragma unroll 4
    for (int i = 0; i < HID / 4; ++i) {
        float4 a = pe[i], b = wp[i];
        s = fmaf(a.x, b.x, s); s = fmaf(a.y, b.y, s);
        s = fmaf(a.z, b.z, s); s = fmaf(a.w, b.w, s);
    }
    int b_ = row >> 11, s_ = row & (SEQ - 1);
    pb[(((size_t)b_ * NHEADS + h) << 11) + s_] = s;
}

// ---------------------------------------------------------------------------
// Pack int32 mask (B,S,S) into bit-mask words: mbits[b][q][k/32]
// Cuts per-head mask traffic 32x (536 MB -> 17 MB total at 23.3 TB/s HBM).
// ---------------------------------------------------------------------------
__global__ __launch_bounds__(256)
void pack_mask_kernel(const int* __restrict__ mask, unsigned int* __restrict__ mbits) {
    int idx = blockIdx.x * blockDim.x + threadIdx.x;      // B*S*(S/32) threads
    const int* p = mask + (size_t)idx * 32;
    unsigned int w = 0;
#pragma unroll
    for (int i = 0; i < 32; ++i) w |= (p[i] != 0 ? 1u : 0u) << i;
    mbits[idx] = w;
}

// ---------------------------------------------------------------------------
// bf16 TN GEMM:  out(M=4096, N=1024) = A(M,K=1024) @ W(N,K)^T, +bias, *premul
// MODE 0: store f32 row-major    MODE 1: store bf16 row-major
// MODE 2: store bf16 transposed-per-head: vT[b][h][d][s]  (for V)
// Block = 256 thr (8 waves); wave -> 16x64 tile; 4 WMMAs / 32-K step.
// ---------------------------------------------------------------------------
template <int MODE>
__global__ __launch_bounds__(256)
void gemm_bf16_kernel(const unsigned short* __restrict__ A,
                      const unsigned short* __restrict__ W,
                      const float* __restrict__ bias,
                      void* __restrict__ out, float premul) {
    const int lane = threadIdx.x & 31;
    const int wid  = threadIdx.x >> 5;
    const int ln   = lane & 15;       // N column / M row within fragment
    const int kh   = lane >> 4;       // K-half selector
    const int m0   = blockIdx.x * 128 + wid * 16;
    const int n0   = blockIdx.y * 64;

    v8f acc[4];
#pragma unroll
    for (int nt = 0; nt < 4; ++nt) acc[nt] = v8f_zero();

    const unsigned short* arow = A + (size_t)(m0 + ln) * HID;

    for (int k = 0; k < HID; k += 32) {
        Frag16 af;
        const unsigned short* ap = arow + k + 8 * kh;
        af.u[0] = *(const uint4*)ap;
        af.u[1] = *(const uint4*)(ap + 16);
#pragma unroll
        for (int nt = 0; nt < 4; ++nt) {
            Frag16 bf;
            const unsigned short* bp = W + (size_t)(n0 + nt * 16 + ln) * HID + k + 8 * kh;
            bf.u[0] = *(const uint4*)bp;
            bf.u[1] = *(const uint4*)(bp + 16);
            acc[nt] = wmma_bf16(af.v, bf.v, acc[nt]);
        }
    }

#pragma unroll
    for (int nt = 0; nt < 4; ++nt) {
        const int ncol = n0 + nt * 16 + ln;
        const float bv = bias[ncol];
#pragma unroll
        for (int r = 0; r < 8; ++r) {
            const int mrow = m0 + 8 * kh + r;            // C layout: row = 8*half + vgpr
            float val = (acc[nt][r] + bv) * premul;
            if (MODE == 0) {
                ((float*)out)[(size_t)mrow * HID + ncol] = val;
            } else if (MODE == 1) {
                ((unsigned short*)out)[(size_t)mrow * HID + ncol] = f2bf(val);
            } else {                                     // V transposed per head
                int b_ = mrow >> 11, s_ = mrow & (SEQ - 1);
                int h_ = ncol >> 6, d_ = ncol & (HDIM - 1);
                ((unsigned short*)out)[((size_t)((b_ * NHEADS + h_) * HDIM + d_) << 11) + s_] =
                    f2bf(val);
            }
        }
    }
}

// ---------------------------------------------------------------------------
// Flash attention: grid (S/64, B*NHEADS), 128 thr (4 waves), wave = 16 q-rows.
// Per 32-key step: 4 score WMMAs + online softmax (DPP reductions) +
// LDS C->A re-layout + 4 PV WMMAs.
// Q pre-scaled by 1/sqrt(hd); pos_bias added per key; mask via packed bits.
// ---------------------------------------------------------------------------
__global__ __launch_bounds__(128)
void attn_kernel(const unsigned short* __restrict__ q,
                 const unsigned short* __restrict__ kbuf,
                 const unsigned short* __restrict__ vT,
                 const float* __restrict__ pos_bias,
                 const unsigned int* __restrict__ mbits,
                 unsigned short* __restrict__ ctx) {
    // LDS P-tile per wave: 16 rows x stride 40 halves (80 B: 16B-aligned rows,
    // start-bank pattern m*20 % 64 -> conflict-free b128 reads).
    __shared__ unsigned short plds[4 * 16 * 40];

    const int lane = threadIdx.x & 31;
    const int wid  = threadIdx.x >> 5;
    const int ln   = lane & 15;
    const int kh   = lane >> 4;
    const int q0   = blockIdx.x * 64 + wid * 16;
    const int bh   = blockIdx.y;
    const int b_   = bh >> 4;
    const int h_   = bh & (NHEADS - 1);

    // Load Q A-fragments (held in registers for whole kernel)
    Frag16 qa[2];
    {
        const unsigned short* qp = q + (size_t)(b_ * SEQ + q0 + ln) * HID + h_ * HDIM;
#pragma unroll
        for (int c = 0; c < 2; ++c) {
            const unsigned short* p = qp + c * 32 + 8 * kh;
            qa[c].u[0] = *(const uint4*)p;
            qa[c].u[1] = *(const uint4*)(p + 16);
        }
    }

    v8f acc[4];
#pragma unroll
    for (int nt = 0; nt < 4; ++nt) acc[nt] = v8f_zero();
    float runm[8], runs[8];
#pragma unroll
    for (int r = 0; r < 8; ++r) { runm[r] = NEG_BIG; runs[r] = 0.0f; }

    unsigned short* myp = plds + wid * (16 * 40);
    const float* pbrow = pos_bias + (size_t)bh * SEQ;
    const unsigned short* krow0 = kbuf + (size_t)(b_ * SEQ) * HID + h_ * HDIM;
    const unsigned short* vbase = vT + ((size_t)(b_ * NHEADS + h_) * HDIM << 11);
    // mask row base for this wave's 8 rows (rows q0 + 8*kh + r)
    const unsigned int* mrow = mbits + (size_t)(b_ * SEQ + q0 + 8 * kh) * (SEQ / 32);

    for (int k0 = 0; k0 < SEQ; k0 += 32) {
        // ---- scores: two 16x16 tiles over 32 keys, K(hd)=64 in 2 chunks ----
        v8f s0 = v8f_zero(), s1 = v8f_zero();
#pragma unroll
        for (int c = 0; c < 2; ++c) {
            Frag16 kf0, kf1;
            const unsigned short* kp0 = krow0 + (size_t)(k0 + ln) * HID + c * 32 + 8 * kh;
            const unsigned short* kp1 = krow0 + (size_t)(k0 + 16 + ln) * HID + c * 32 + 8 * kh;
            kf0.u[0] = *(const uint4*)kp0; kf0.u[1] = *(const uint4*)(kp0 + 16);
            kf1.u[0] = *(const uint4*)kp1; kf1.u[1] = *(const uint4*)(kp1 + 16);
            s0 = wmma_bf16(qa[c].v, kf0.v, s0);
            s1 = wmma_bf16(qa[c].v, kf1.v, s1);
        }

        const float pb0 = pbrow[k0 + ln];
        const float pb1 = pbrow[k0 + 16 + ln];

        // mask words for all 8 rows up-front (clause-able, overlaps WMMAs)
        unsigned int mw[8];
#pragma unroll
        for (int r = 0; r < 8; ++r) mw[r] = mrow[(size_t)r * (SEQ / 32) + (k0 >> 5)];

        // ---- masked online softmax, per fragment row r (DPP reductions) ----
        float corr[8];
#pragma unroll
        for (int r = 0; r < 8; ++r) {
            float a0 = ((mw[r] >> ln) & 1u)        ? s0[r] + pb0 : NEG_BIG;
            float a1 = ((mw[r] >> (16 + ln)) & 1u) ? s1[r] + pb1 : NEG_BIG;
            float tm = row_max16(fmaxf(a0, a1));
            const float nm = fmaxf(runm[r], tm);
            corr[r] = exp2f((runm[r] - nm) * LOG2E);
            runm[r] = nm;
            float p0 = (a0 < NEG_THR) ? 0.0f : exp2f((a0 - nm) * LOG2E);
            float p1 = (a1 < NEG_THR) ? 0.0f : exp2f((a1 - nm) * LOG2E);
            float rs = row_sum16(p0 + p1);
            runs[r] = runs[r] * corr[r] + rs;
            // C-layout -> LDS (16 x 32 bf16 tile, stride 40)
            myp[(8 * kh + r) * 40 + ln]      = f2bf(p0);
            myp[(8 * kh + r) * 40 + 16 + ln] = f2bf(p1);
        }

        // rescale running output accumulators
#pragma unroll
        for (int nt = 0; nt < 4; ++nt)
#pragma unroll
            for (int r = 0; r < 8; ++r) acc[nt][r] *= corr[r];

        __builtin_amdgcn_wave_barrier();   // order intra-wave LDS store -> load

        // reload P as A-fragment (keys = K dimension)
        Frag16 pa;
        {
            const unsigned short* pp = myp + ln * 40 + 8 * kh;
            pa.u[0] = *(const uint4*)pp;
            pa.u[1] = *(const uint4*)(pp + 16);
        }
        __builtin_amdgcn_wave_barrier();

        // ---- PV: 4 B-fragments from transposed V (contiguous along keys) ----
#pragma unroll
        for (int nt = 0; nt < 4; ++nt) {
            Frag16 vf;
            const unsigned short* vp = vbase + ((size_t)(nt * 16 + ln) << 11) + k0 + 8 * kh;
            vf.u[0] = *(const uint4*)vp;
            vf.u[1] = *(const uint4*)(vp + 16);
            acc[nt] = wmma_bf16(pa.v, vf.v, acc[nt]);
        }
    }

    // ---- finalize: divide by softmax sum, store context (bf16) ----
#pragma unroll
    for (int r = 0; r < 8; ++r) runs[r] = 1.0f / runs[r];
#pragma unroll
    for (int nt = 0; nt < 4; ++nt)
#pragma unroll
        for (int r = 0; r < 8; ++r) {
            const int qrow = q0 + 8 * kh + r;
            ctx[(size_t)(b_ * SEQ + qrow) * HID + h_ * HDIM + nt * 16 + ln] =
                f2bf(acc[nt][r] * runs[r]);
        }
}

// ---------------------------------------------------------------------------
// Host-side launch
// ---------------------------------------------------------------------------
extern "C" void kernel_launch(void* const* d_in, const int* in_sizes, int n_in,
                              void* d_out, int out_size, void* d_ws, size_t ws_size,
                              hipStream_t stream) {
    (void)in_sizes; (void)n_in; (void)out_size; (void)ws_size;

    const float* x       = (const float*)d_in[0];
    const float* pos_emb = (const float*)d_in[1];
    const int*   mask    = (const int*)  d_in[2];
    const float* wq      = (const float*)d_in[3];
    const float* bq      = (const float*)d_in[4];
    const float* wk      = (const float*)d_in[5];
    const float* bk      = (const float*)d_in[6];
    const float* wv      = (const float*)d_in[7];
    const float* bv      = (const float*)d_in[8];
    const float* wo      = (const float*)d_in[9];
    const float* bo      = (const float*)d_in[10];
    const float* w_pos   = (const float*)d_in[11];

    // Workspace layout (bytes), all offsets 4KB-aligned
    char* ws = (char*)d_ws;
    const size_t XB   = (size_t)MTOT * HID * 2;         // 8 MB bf16 activations
    const size_t WB   = (size_t)HID * HID * 2;          // 2 MB bf16 weights
    unsigned short* xb  = (unsigned short*)(ws);
    unsigned short* wqb = (unsigned short*)(ws + XB);
    unsigned short* wkb = (unsigned short*)(ws + XB + WB);
    unsigned short* wvb = (unsigned short*)(ws + XB + 2 * WB);
    unsigned short* wob = (unsigned short*)(ws + XB + 3 * WB);
    unsigned short* qb  = (unsigned short*)(ws + XB + 4 * WB);
    unsigned short* kb  = (unsigned short*)(ws + 2 * XB + 4 * WB);
    unsigned short* vT  = (unsigned short*)(ws + 3 * XB + 4 * WB);
    unsigned short* ctx = (unsigned short*)(ws + 4 * XB + 4 * WB);
    float*        pbias = (float*)       (ws + 5 * XB + 4 * WB);
    unsigned int* mbits = (unsigned int*)(ws + 5 * XB + 4 * WB +
                                          (size_t)BATCH * NHEADS * SEQ * 4);

    // 1) fp32 -> bf16 conversions
    cvt_bf16_kernel<<<(MTOT * HID / 4 + 255) / 256, 256, 0, stream>>>(x,  xb,  MTOT * HID / 4);
    cvt_bf16_kernel<<<(HID * HID / 4 + 255) / 256, 256, 0, stream>>>(wq, wqb, HID * HID / 4);
    cvt_bf16_kernel<<<(HID * HID / 4 + 255) / 256, 256, 0, stream>>>(wk, wkb, HID * HID / 4);
    cvt_bf16_kernel<<<(HID * HID / 4 + 255) / 256, 256, 0, stream>>>(wv, wvb, HID * HID / 4);
    cvt_bf16_kernel<<<(HID * HID / 4 + 255) / 256, 256, 0, stream>>>(wo, wob, HID * HID / 4);

    // 2) positional bias + packed mask
    pos_bias_kernel<<<(BATCH * SEQ * NHEADS) / 256, 256, 0, stream>>>(pos_emb, w_pos, pbias);
    pack_mask_kernel<<<(BATCH * SEQ * (SEQ / 32)) / 256, 256, 0, stream>>>(mask, mbits);

    // 3) QKV projections (WMMA GEMMs). Q pre-scaled by 1/sqrt(hd); V transposed.
    dim3 ggrid(MTOT / 128, HID / 64);
    gemm_bf16_kernel<1><<<ggrid, 256, 0, stream>>>(xb, wqb, bq, qb, QK_SCALE);
    gemm_bf16_kernel<1><<<ggrid, 256, 0, stream>>>(xb, wkb, bk, kb, 1.0f);
    gemm_bf16_kernel<2><<<ggrid, 256, 0, stream>>>(xb, wvb, bv, vT, 1.0f);

    // 4) flash attention
    dim3 agrid(SEQ / 64, BATCH * NHEADS);
    attn_kernel<<<agrid, 128, 0, stream>>>(qb, kb, vT, pbias, mbits, ctx);

    // 5) output projection -> fp32 d_out
    gemm_bf16_kernel<0><<<ggrid, 256, 0, stream>>>(ctx, wob, bo, d_out, 1.0f);
}
